// RnCLoss_68805376081954
// MI455X (gfx1250) — compile-verified, matrix operand order checked by reading, not run on hardware
//
#include <hip/hip_runtime.h>

typedef __attribute__((ext_vector_type(16))) _Float16 v16h;
typedef __attribute__((ext_vector_type(8)))  _Float16 v8h;
typedef __attribute__((ext_vector_type(8)))  float    v8f;

#define N   512
#define TPB 256

// ---------------------------------------------------------------------------
// Kernel 1: L2-normalize each row, emit FP16 copy for the WMMA Gram kernel.
// One block per row; 256 threads handle 512 elements (2 each).
// ---------------------------------------------------------------------------
__global__ void normalize_f16_kernel(const float* __restrict__ feats,
                                     _Float16* __restrict__ fh) {
  __shared__ float red[TPB];
  const int row = blockIdx.x;
  const int tid = threadIdx.x;
  const float* r = feats + row * N;
  float x0 = r[tid];
  float x1 = r[tid + TPB];
  red[tid] = x0 * x0 + x1 * x1;
  __syncthreads();
  for (int s = TPB / 2; s > 0; s >>= 1) {
    if (tid < s) red[tid] += red[tid + s];
    __syncthreads();
  }
  const float rn = rsqrtf(red[0]);
  fh[row * N + tid]       = (_Float16)(x0 * rn);
  fh[row * N + tid + TPB] = (_Float16)(x1 * rn);
}

// ---------------------------------------------------------------------------
// Kernel 2: Gram matrix via v_wmma_f32_16x16x32_f16, fused epilogue:
//   g = f_i . f_j ;  sq = max(2 - 2g, 0) ;  dist = sqrt(sq)
//   elog[i,j] = exp(dist / T), T = 2
// One wave (32 lanes) per 16x16 output tile; 8 waves / block.
// A-operand: lane l holds row (tm + (l&15)), K-chunks [h*8,h*8+8) and
// [16+h*8, 16+h*8+8) with h = l>>4 (per ISA 16-bit A layout). For B = f^T the
// per-lane load pattern is identical with the column tile's rows.
// ---------------------------------------------------------------------------
__global__ void gram_explog_kernel(const _Float16* __restrict__ fh,
                                   float* __restrict__ elog) {
  const int wave = threadIdx.x >> 5;
  const int lane = threadIdx.x & 31;
  const int tile = blockIdx.x * 8 + wave;
  const int tm = (tile >> 5) * 16;   // 32 tiles per dimension
  const int tn = (tile & 31) * 16;
  const int r = lane & 15;
  const int h = lane >> 4;

  const _Float16* arow = fh + (tm + r) * N;
  const _Float16* brow = fh + (tn + r) * N;

  v8f acc = {};
#pragma unroll
  for (int kb = 0; kb < N; kb += 32) {
    v8h a0 = *(const v8h*)(arow + kb + h * 8);
    v8h a1 = *(const v8h*)(arow + kb + 16 + h * 8);
    v8h b0 = *(const v8h*)(brow + kb + h * 8);
    v8h b1 = *(const v8h*)(brow + kb + 16 + h * 8);
    v16h a = __builtin_shufflevector(a0, a1, 0, 1, 2, 3, 4, 5, 6, 7,
                                             8, 9, 10, 11, 12, 13, 14, 15);
    v16h b = __builtin_shufflevector(b0, b1, 0, 1, 2, 3, 4, 5, 6, 7,
                                             8, 9, 10, 11, 12, 13, 14, 15);
    acc = __builtin_amdgcn_wmma_f32_16x16x32_f16(
        /*neg_a=*/false, a, /*neg_b=*/false, b,
        /*c_mod=*/(short)0, acc, /*reuse_a=*/false, /*reuse_b=*/false);
  }

  // C/D layout: acc[g] -> (M = g + 8*h, Ncol = r)
#pragma unroll
  for (int g = 0; g < 8; ++g) {
    float gv = acc[g];
    float sq = 2.0f - 2.0f * gv;
    sq = sq > 0.0f ? sq : 0.0f;
    float dist = sqrtf(sq);
    float logit = dist * 0.5f;                 // / T, T = 2
    elog[(tm + g + 8 * h) * N + tn + r] = __expf(logit);
  }
}

// ---------------------------------------------------------------------------
// Kernel 3: per-anchor-row RnC denominator + row loss, fully LDS-resident.
// denom[k] = sum_{j != i, ld[j] >= ld[k]} e[j]   (j == k included: ld[k]>=ld[k])
// row_loss[i] = sum_{k != i} ( logit[i,k] - log(denom[k]) ),
// with logit[i,k] = log(e[k]). Diagonal excluded by zeroing e[i] and skipping
// k == i. Deterministic block-tree reduction.
// ---------------------------------------------------------------------------
__global__ void denom_loss_kernel(const float* __restrict__ elog,
                                  const float* __restrict__ labels,
                                  float* __restrict__ rowloss) {
  __shared__ float e[N];
  __shared__ float ld[N];
  __shared__ float red[TPB];
  const int i = blockIdx.x;
  const int tid = threadIdx.x;
  const float li = labels[i];
  for (int j = tid; j < N; j += TPB) {
    e[j]  = elog[i * N + j];
    ld[j] = fabsf(li - labels[j]);
  }
  __syncthreads();
  if (tid == 0) e[i] = 0.0f;   // drop diagonal term from every denominator
  __syncthreads();

  float local = 0.0f;
  for (int k = tid; k < N; k += TPB) {
    if (k == i) continue;
    const float t = ld[k];
    float denom = 0.0f;
    for (int j = 0; j < N; ++j) {
      denom += (ld[j] >= t) ? e[j] : 0.0f;
    }
    local += logf(e[k]) - logf(denom);
  }
  red[tid] = local;
  __syncthreads();
  for (int s = TPB / 2; s > 0; s >>= 1) {
    if (tid < s) red[tid] += red[tid + s];
    __syncthreads();
  }
  if (tid == 0) rowloss[i] = red[0];
}

// ---------------------------------------------------------------------------
// Kernel 4: deterministic final reduction over the 512 row losses.
// ---------------------------------------------------------------------------
__global__ void final_reduce_kernel(const float* __restrict__ rowloss,
                                    float* __restrict__ out) {
  __shared__ float red[TPB];
  const int tid = threadIdx.x;
  red[tid] = rowloss[tid] + rowloss[tid + TPB];
  __syncthreads();
  for (int s = TPB / 2; s > 0; s >>= 1) {
    if (tid < s) red[tid] += red[tid + s];
    __syncthreads();
  }
  if (tid == 0) out[0] = -red[0] / (512.0f * 511.0f);
}

// ---------------------------------------------------------------------------
// Workspace layout:
//   fh      : N*N _Float16  (512 KB)
//   elog    : N*N float     (  1 MB)
//   rowloss : N   float     (  2 KB)
// ---------------------------------------------------------------------------
extern "C" void kernel_launch(void* const* d_in, const int* in_sizes, int n_in,
                              void* d_out, int out_size, void* d_ws, size_t ws_size,
                              hipStream_t stream) {
  const float* feats  = (const float*)d_in[0];
  const float* labels = (const float*)d_in[1];
  float* out = (float*)d_out;

  char* ws = (char*)d_ws;
  _Float16* fh   = (_Float16*)ws;
  float*   elog  = (float*)(ws + (size_t)N * N * sizeof(_Float16));
  float* rowloss = (float*)(ws + (size_t)N * N * sizeof(_Float16)
                               + (size_t)N * N * sizeof(float));

  normalize_f16_kernel<<<N, TPB, 0, stream>>>(feats, fh);
  // 32x32 = 1024 tiles, 8 waves (tiles) per 256-thread block -> 128 blocks
  gram_explog_kernel<<<128, TPB, 0, stream>>>(fh, elog);
  denom_loss_kernel<<<N, TPB, 0, stream>>>(elog, labels, rowloss);
  final_reduce_kernel<<<1, TPB, 0, stream>>>(rowloss, out);
}
